// DotProductAttention_32882269618187
// MI455X (gfx1250) — matrix operand order repaired and checked by reading, not compiled, not run on hardware
//
#include <hip/hip_runtime.h>

#define BB 64
#define NQS 1024
#define NKS 1024
#define DD  128

typedef __attribute__((ext_vector_type(16))) __bf16 v16bf;
typedef __attribute__((ext_vector_type(8)))  __bf16 v8bf;
typedef __attribute__((ext_vector_type(8)))  float  v8f;

// raw hardware exp2 (v_exp_f32): flush-to-zero on the denormal tail is fine here
// (inputs are <= 0; masked scores are -1e6 and must map to 0 anyway).
__device__ __forceinline__ float fast_exp2(float x) {
  return __builtin_amdgcn_exp2f(x);
}

// ---------------- prepass 1: K (f32) -> Kb (bf16, same [B][NK][D] layout) ----
__global__ __launch_bounds__(256)
void cvt_bf16_kernel(const float* __restrict__ in, __bf16* __restrict__ out, int n8) {
  int i = blockIdx.x * blockDim.x + threadIdx.x;
  if (i >= n8) return;
  const float4* p = (const float4*)in + 2 * (size_t)i;
  float4 a = p[0], b = p[1];
  v8bf o;
  o[0]=(__bf16)a.x; o[1]=(__bf16)a.y; o[2]=(__bf16)a.z; o[3]=(__bf16)a.w;
  o[4]=(__bf16)b.x; o[5]=(__bf16)b.y; o[6]=(__bf16)b.z; o[7]=(__bf16)b.w;
  ((v8bf*)out)[i] = o;
}

// ---------------- prepass 2: V [B][NK][D] f32 -> Vt [B][D][NK] bf16 ----------
__global__ __launch_bounds__(256)
void transpose_v_kernel(const float* __restrict__ V, __bf16* __restrict__ Vt) {
  __shared__ float tile[32][33];
  const int b  = blockIdx.z;
  const int kt = blockIdx.x * 32;   // key tile
  const int dt = blockIdx.y * 32;   // d tile
  const int tx = threadIdx.x, ty = threadIdx.y;      // (32, 8)
  const float* src = V + (size_t)b * NKS * DD;
  #pragma unroll
  for (int j = 0; j < 32; j += 8)
    tile[ty + j][tx] = src[(size_t)(kt + ty + j) * DD + dt + tx];
  __syncthreads();
  __bf16* dst = Vt + (size_t)b * DD * NKS;
  #pragma unroll
  for (int j = 0; j < 32; j += 8)
    dst[(size_t)(dt + ty + j) * NKS + kt + tx] = (__bf16)tile[tx][ty + j];
}

// ---------------- main flash-attention kernel --------------------------------
// One wave = 16 queries.  S^T = K * Q^T  (keys as M, queries as N),
// O^T accumulated as V^T * P^T.  Output transposed back through LDS.
//
// Q fragments are pre-scaled by (1/sqrt(D))*log2(e) so scores live in the
// log2 domain: p = exp2(s - m).
//
// The PV contraction uses a permuted logical key order {0..7,16..23,8..15,24..31}
// applied identically to the V^T A-operand addressing and the P^T B-operand, so
// P^T is built purely from lane-local WMMA accumulator values (no shuffles) and
// the V^T fragment is two 16-byte loads per lane (keys +0..7 and +16..23).
__global__ __launch_bounds__(256)
void fa_fwd_kernel(const float* __restrict__ Q,
                   const __bf16* __restrict__ Kb,
                   const __bf16* __restrict__ Vt,
                   const int* __restrict__ VL,
                   float* __restrict__ O) {
  __shared__ __align__(16) float smem[8][16 * 132];   // per-wave O transpose buffer

  const int lane = threadIdx.x & 31;
  const int wave = threadIdx.x >> 5;
  const int hi   = lane >> 4;          // lane half (0/1)
  const int l16  = lane & 15;
  const int b     = blockIdx.x >> 3;                    // 8 q-blocks / batch
  const int qbase = (blockIdx.x & 7) * 128 + wave * 16; // this wave's 16 queries
  const int valid = VL[b];
  const int nb32  = ((valid + 31) >> 5) << 5;           // skip fully-masked key blocks

  // (1/sqrt(128)) * log2(e): folded into Q once.
  const float QSCALE = 0.08838834764831845f * 1.4426950408889634f;

  // ---- Q as B-operand: lane l<16 -> query l; element e -> d = c*32 + hi*16 + e
  v16bf qf[4];
  {
    const float* qrow = Q + ((size_t)b * NQS + qbase + l16) * DD;
    #pragma unroll
    for (int c = 0; c < 4; ++c) {
      float t[16];
      const float4* s = (const float4*)(qrow + c * 32 + hi * 16);
      *(float4*)&t[0]  = s[0];
      *(float4*)&t[4]  = s[1];
      *(float4*)&t[8]  = s[2];
      *(float4*)&t[12] = s[3];
      #pragma unroll
      for (int e = 0; e < 16; ++e) qf[c][e] = (__bf16)(t[e] * QSCALE);
    }
  }

  v8f acc[8];                                  // O^T: 8 d-tiles of 16x16 f32
  #pragma unroll
  for (int t = 0; t < 8; ++t)
    #pragma unroll
    for (int r = 0; r < 8; ++r) acc[t][r] = 0.0f;

  float mrun = -1e30f, lrun = 0.0f;

  const __bf16* kbb = Kb + (size_t)b * NKS * DD;
  const __bf16* vtb = Vt + (size_t)b * DD * NKS;

  for (int kb = 0; kb < nb32; kb += 32) {
    // speculative next-block prefetch (safe past the end; stays inside d_ws)
    __builtin_prefetch(kbb + (size_t)(kb + 32 + l16) * DD, 0, 1);
    __builtin_prefetch(vtb + (size_t)l16 * NKS + kb + 32, 0, 1);

    // ---- S^T = K * Q^T : two 16-key tiles, 4 d-chunks of 32 each
    v8f s0 = {0,0,0,0,0,0,0,0}, s1 = {0,0,0,0,0,0,0,0};
    #pragma unroll
    for (int c = 0; c < 4; ++c) {
      const int cb = c * 32 + hi * 16;     // A-layout: {0..15} / {16..31}
      const __bf16* kr0 = kbb + (size_t)(kb + l16) * DD + cb;
      const __bf16* kr1 = kr0 + 16 * DD;
      v8bf a0l = *(const v8bf*)kr0;
      v8bf a0h = *(const v8bf*)(kr0 + 8);
      v8bf a1l = *(const v8bf*)kr1;
      v8bf a1h = *(const v8bf*)(kr1 + 8);
      v16bf a0, a1;
      #pragma unroll
      for (int e = 0; e < 8; ++e) {
        a0[e] = a0l[e]; a0[e + 8] = a0h[e];
        a1[e] = a1l[e]; a1[e + 8] = a1h[e];
      }
      s0 = __builtin_amdgcn_wmma_f32_16x16x32_bf16(false, a0, false, qf[c], (short)0, s0, false, false);
      s1 = __builtin_amdgcn_wmma_f32_16x16x32_bf16(false, a1, false, qf[c], (short)0, s1, false, false);
    }

    // ---- key-length mask (only the last block can be partial); log2 domain
    if (kb + 32 > valid) {
      #pragma unroll
      for (int r = 0; r < 8; ++r) {
        if (kb + r + 8 * hi      >= valid) s0[r] = -1.0e6f;
        if (kb + r + 8 * hi + 16 >= valid) s1[r] = -1.0e6f;
      }
    }

    // ---- online softmax: each lane holds 16 scores of its query; partner lane
    //      (lane^16) holds the other 16 of this 32-key block.
    float mloc = s0[0];
    #pragma unroll
    for (int r = 1; r < 8; ++r) mloc = fmaxf(mloc, s0[r]);
    #pragma unroll
    for (int r = 0; r < 8; ++r) mloc = fmaxf(mloc, s1[r]);
    const float mblk = fmaxf(mloc, __shfl_xor(mloc, 16, 32));
    const float mnew = fmaxf(mrun, mblk);

    // rescale running state only when some lane's max actually increased
    // (resc == 1.0 exactly whenever this is skipped)
    if (__ballot(mblk > mrun)) {
      const float resc = fast_exp2(mrun - mnew);
      lrun *= resc;
      #pragma unroll
      for (int t = 0; t < 8; ++t)
        #pragma unroll
        for (int r = 0; r < 8; ++r) acc[t][r] *= resc;
    }
    mrun = mnew;

    float p0[8], p1[8], psum = 0.0f;
    #pragma unroll
    for (int r = 0; r < 8; ++r) {
      p0[r] = fast_exp2(s0[r] - mnew);
      p1[r] = fast_exp2(s1[r] - mnew);
      psum += p0[r] + p1[r];
    }
    lrun += psum;                  // per-lane partial sum; combined at the end

    // ---- P^T B-operand in permuted key order: logical K (per lane half) is
    //      {0..7,16..23} / {8..15,24..31} physical -> exactly this lane's local
    //      accumulator values.  No cross-lane exchange, no selects.
    v16bf pf;
    #pragma unroll
    for (int e = 0; e < 8; ++e) {
      pf[e]     = (__bf16)p0[e];
      pf[e + 8] = (__bf16)p1[e];
    }

    // ---- O^T += V^T * P^T  (8 d-tiles).  Same permuted key order on the
    //      A side: per-lane keys kb + hi*8 + {0..7,16..23}.
    #pragma unroll
    for (int t = 0; t < 8; ++t) {
      const __bf16* vr = vtb + (size_t)(t * 16 + l16) * NKS + kb + hi * 8;
      v8bf vl_ = *(const v8bf*)vr;
      v8bf vh_ = *(const v8bf*)(vr + 16);
      v16bf av;
      #pragma unroll
      for (int e = 0; e < 8; ++e) { av[e] = vl_[e]; av[e + 8] = vh_[e]; }
      acc[t] = __builtin_amdgcn_wmma_f32_16x16x32_bf16(false, av, false, pf, (short)0, acc[t], false, false);
    }
  }

  // ---- normalize and transpose O^T -> O through LDS, coalesced b128 stores
  const float ltot = lrun + __shfl_xor(lrun, 16, 32);
  const float inv  = 1.0f / ltot;
  #pragma unroll
  for (int t = 0; t < 8; ++t)
    #pragma unroll
    for (int r = 0; r < 8; ++r)
      smem[wave][l16 * 132 + t * 16 + r + 8 * hi] = acc[t][r] * inv;
  __syncthreads();

  float* od = O + ((size_t)b * NQS + qbase) * DD;
  #pragma unroll
  for (int i = 0; i < 16; ++i) {
    float4 v = *(const float4*)&smem[wave][i * 132 + lane * 4];
    *(float4*)(od + (size_t)i * DD + lane * 4) = v;
  }
}

// ---------------------------------------------------------------------------
extern "C" void kernel_launch(void* const* d_in, const int* in_sizes, int n_in,
                              void* d_out, int out_size, void* d_ws, size_t ws_size,
                              hipStream_t stream) {
  (void)in_sizes; (void)n_in; (void)out_size; (void)ws_size;
  const float* Q  = (const float*)d_in[0];
  const float* K  = (const float*)d_in[1];
  const float* V  = (const float*)d_in[2];
  const int*   VL = (const int*)d_in[3];
  float* O = (float*)d_out;

  const size_t elems = (size_t)BB * NKS * DD;          // 8.4M
  __bf16* Kb = (__bf16*)d_ws;
  __bf16* Vt = (__bf16*)((char*)d_ws + elems * sizeof(__bf16));

  const int n8 = (int)(elems / 8);
  cvt_bf16_kernel<<<(n8 + 255) / 256, 256, 0, stream>>>(K, Kb, n8);
  transpose_v_kernel<<<dim3(NKS / 32, DD / 32, BB), dim3(32, 8), 0, stream>>>(V, Vt);
  fa_fwd_kernel<<<BB * 8, 256, 0, stream>>>(Q, Kb, Vt, VL, O);
}
